// RIPGeoContrast_21801253994575
// MI455X (gfx1250) — matrix operand order repaired and checked by reading, not compiled, not run on hardware
//
#include <hip/hip_runtime.h>
#include <hip/hip_bf16.h>

// ---------------------------------------------------------------------------
// RIPGeoContrast pipeline for MI455X (gfx1250, wave32, WMMA).
// Heavy math (all convs + adj) runs through a 128x128x32-tile WMMA GEMM
// (v_wmma_f32_16x16x32_f16) with double-buffered GLOBAL_LOAD_ASYNC_TO_LDS
// staging (ASYNCcnt). Selection stages use per-row LDS radix-select.
// ---------------------------------------------------------------------------

typedef __attribute__((ext_vector_type(16))) _Float16 v16h;
typedef __attribute__((ext_vector_type(8)))  _Float16 v8h;
typedef __attribute__((ext_vector_type(8)))  float    v8f;

namespace {

constexpr int   N_TOT   = 8192;
constexpr int   N_LM    = 4096;
constexpr int   FEAT    = 31;
constexpr int   DIMZ    = 512;
constexpr int   MEMD    = 30;
constexpr float SHRINK  = 0.0025f;
constexpr int   K_ADJ   = 2048;   // n_lm // 2
constexpr int   K_KNN   = 819;    // int(0.1 * 8192)

// ------------------------------ GEMM (WMMA) --------------------------------
constexpr int BM = 128, BN = 128, BK = 32;
constexpr int A_LD = BK + 8;   // f16 stride, bank-conflict pad (16B aligned)
constexpr int B_LD = BK + 8;   // Bs stored [n][k]

__device__ inline unsigned ordKey(float f) {
  unsigned u = __float_as_uint(f);
  return u ^ ((u & 0x80000000u) ? 0xFFFFFFFFu : 0x80000000u);
}

// CDNA5 direct memory->LDS async copy (tracked by ASYNCcnt).
__device__ __forceinline__ void async_b128(unsigned ldsOff, const void* g) {
  unsigned long long ga = (unsigned long long)(uintptr_t)g;
  asm volatile("global_load_async_to_lds_b128 %0, %1, off"
               :: "v"(ldsOff), "v"(ga) : "memory");
}
__device__ __forceinline__ void wait_async0() {
  asm volatile("s_wait_asynccnt 0" ::: "memory");
}

// C(M x Nc) = A(M x Kp) * Bt^T.  A row-major lda=Kp.
// Bt row-major (Nrows x Kp): holds B^T (weights pre-transposed / feature NT).
// M multiple of 128, Kp multiple of 32; Bt has >= round_up(Nc,128) rows.
__global__ __launch_bounds__(256)
void gemm_f16_wmma(const _Float16* __restrict__ A, int lda,
                   const _Float16* __restrict__ Bt, int ldb,
                   float* __restrict__ C, int ldc,
                   int M, int Ncols, int Kp)
{
  __shared__ _Float16 As[2 * BM * A_LD];
  __shared__ _Float16 Bs[2 * BN * B_LD];

  const int tid  = threadIdx.x;
  const int lane = tid & 31;
  const int wave = tid >> 5;          // 8 waves
  const int wm   = wave & 3;          // 4 wave-rows  (32 rows each)
  const int wn   = wave >> 2;         // 2 wave-cols  (64 cols each)
  const int m0   = blockIdx.y * BM;
  const int n0   = blockIdx.x * BN;
  const int r    = lane & 15;
  const int h    = lane >> 4;

  // LDS byte offsets (low 32 bits of generic address = LDS offset)
  const unsigned asBase = (unsigned)(uintptr_t)&As[0];
  const unsigned bsBase = (unsigned)(uintptr_t)&Bs[0];

  // per-thread staging coords: 512 8-half chunks per 128x32 tile, 2 per thread
  const int sr0 = tid >> 2,           sc0 = (tid & 3) * 8;
  const int sr1 = (tid + 256) >> 2,   sc1 = ((tid + 256) & 3) * 8;

  auto issue = [&](int k0, int buf) {
    unsigned aO = asBase + (unsigned)(buf * BM * A_LD * 2);
    unsigned bO = bsBase + (unsigned)(buf * BN * B_LD * 2);
    async_b128(aO + (unsigned)((sr0 * A_LD + sc0) * 2),
               A + (size_t)(m0 + sr0) * lda + k0 + sc0);
    async_b128(aO + (unsigned)((sr1 * A_LD + sc1) * 2),
               A + (size_t)(m0 + sr1) * lda + k0 + sc1);
    async_b128(bO + (unsigned)((sr0 * B_LD + sc0) * 2),
               Bt + (size_t)(n0 + sr0) * ldb + k0 + sc0);
    async_b128(bO + (unsigned)((sr1 * B_LD + sc1) * 2),
               Bt + (size_t)(n0 + sr1) * ldb + k0 + sc1);
  };

  v8f vz = {};
  v8f acc[2][4];
#pragma unroll
  for (int i = 0; i < 2; ++i)
#pragma unroll
    for (int j = 0; j < 4; ++j) acc[i][j] = vz;

  issue(0, 0);
  wait_async0();
  __syncthreads();

  int buf = 0;
  for (int k0 = 0; k0 < Kp; k0 += BK) {
    // overlap: start async staging of the next K-slab into the other buffer
    if (k0 + BK < Kp) issue(k0 + BK, buf ^ 1);
    if (k0 + 2 * BK < Kp) {   // warm GL2 for the slab after next
      __builtin_prefetch(A  + (size_t)(m0 + sr0) * lda + k0 + 2 * BK, 0, 3);
      __builtin_prefetch(Bt + (size_t)(n0 + sr0) * ldb + k0 + 2 * BK, 0, 3);
    }

    const _Float16* Ab = As + buf * BM * A_LD;
    const _Float16* Bb = Bs + buf * BN * B_LD;

    // fragments per ISA 7.12.2 (16-bit A 16x32; B mirrored via [n][k] tile)
    v16h af[2], bf[4];
#pragma unroll
    for (int i = 0; i < 2; ++i) {
      const _Float16* p = Ab + (wm * 32 + i * 16 + r) * A_LD + h * 8;
      v8h lo = *(const v8h*)p;
      v8h hi = *(const v8h*)(p + 16);
#pragma unroll
      for (int e = 0; e < 8; ++e) { af[i][e] = lo[e]; af[i][e + 8] = hi[e]; }
    }
#pragma unroll
    for (int j = 0; j < 4; ++j) {
      const _Float16* p = Bb + (wn * 64 + j * 16 + r) * B_LD + h * 8;
      v8h lo = *(const v8h*)p;
      v8h hi = *(const v8h*)(p + 16);
#pragma unroll
      for (int e = 0; e < 8; ++e) { bf[j][e] = lo[e]; bf[j][e + 8] = hi[e]; }
    }

#pragma unroll
    for (int i = 0; i < 2; ++i)
#pragma unroll
      for (int j = 0; j < 4; ++j)
        acc[i][j] = __builtin_amdgcn_wmma_f32_16x16x32_f16(
            false, af[i], false, bf[j], (short)0, acc[i][j], false, false);

    wait_async0();       // own async copies into buf^1 have landed
    __syncthreads();     // all waves done reading buf + all asyncs visible
    buf ^= 1;
  }

  // store C: lane = col (l&15), vgpr rr -> row (l>>4)*8 + rr
#pragma unroll
  for (int i = 0; i < 2; ++i) {
#pragma unroll
    for (int j = 0; j < 4; ++j) {
      int col = n0 + wn * 64 + j * 16 + r;
      if (col < Ncols) {
        int mrow = m0 + wm * 32 + i * 16 + h * 8;
#pragma unroll
        for (int rr = 0; rr < 8; ++rr)
          C[(size_t)(mrow + rr) * ldc + col] = acc[i][j][rr];
      }
    }
  }
}

// ---------------------------- support kernels ------------------------------

// Build X f16: (8192 rows x 32), row n = [feat(31), delay]
__global__ void pack_x_k(const float* __restrict__ lmX, const float* __restrict__ lmD,
                         const float* __restrict__ tgX, const float* __restrict__ tgD,
                         _Float16* __restrict__ out)
{
  int idx = blockIdx.x * 256 + threadIdx.x;
  if (idx >= N_TOT * 32) return;
  int l = idx & 31, n = idx >> 5;
  float v;
  if (n < N_LM) v = (l < FEAT) ? lmX[n * FEAT + l] : lmD[n];
  else { int m = n - N_LM; v = (l < FEAT) ? tgX[m * FEAT + l] : tgD[m]; }
  out[idx] = (_Float16)v;
}

// act (N, Lin, Cin) f16 NLC -> col (N*Lout x Kp) f16, k = ci*KS + kk
__global__ void im2col_k(const _Float16* __restrict__ act, _Float16* __restrict__ col,
                         int Lin, int Cin, int Lout, int KS, int stride, int pad, int Kp)
{
  size_t idx = (size_t)blockIdx.x * 256 + threadIdx.x;
  size_t total = (size_t)N_TOT * Lout * Kp;
  if (idx >= total) return;
  int k = (int)(idx % Kp);
  size_t row = idx / Kp;
  int lo = (int)(row % Lout);
  int n  = (int)(row / Lout);
  _Float16 v = (_Float16)0.0f;
  if (k < Cin * KS) {
    int ci = k / KS, kk = k % KS;
    int li = lo * stride - pad + kk;
    if (li >= 0 && li < Lin) v = act[((size_t)n * Lin + li) * Cin + ci];
  }
  col[idx] = v;
}

// w (Cout, Cin, KS) f32 -> wT (Np x Kp) f16 (transposed, zero-padded), so the
// GEMM consumes it as B^T with contiguous 16B async-to-LDS staging.
__global__ void repack_w_k(const float* __restrict__ wsrc, _Float16* __restrict__ out,
                           int Cin, int KS, int Cout, int Kp, int Np)
{
  int idx = blockIdx.x * 256 + threadIdx.x;
  if (idx >= Np * Kp) return;
  int k = idx % Kp, co = idx / Kp;
  float v = 0.f;
  if (co < Cout && k < Cin * KS)
    v = wsrc[((size_t)co * Cin + k / KS) * KS + (k % KS)];
  out[idx] = (_Float16)v;
}

// per-channel mean/var over (M x C) f32, fold g,b into scale/shift
__global__ __launch_bounds__(256)
void bn_stats_k(const float* __restrict__ x, const float* __restrict__ g,
                const float* __restrict__ b, float* __restrict__ scale,
                float* __restrict__ shift, int M, int C)
{
  __shared__ float s_sum[256], s_sq[256];
  int c = blockIdx.x;
  float sum = 0.f, sq = 0.f;
  for (int i = threadIdx.x; i < M; i += 256) {
    float v = x[(size_t)i * C + c];
    sum += v; sq += v * v;
  }
  s_sum[threadIdx.x] = sum; s_sq[threadIdx.x] = sq;
  __syncthreads();
  for (int off = 128; off > 0; off >>= 1) {
    if (threadIdx.x < off) {
      s_sum[threadIdx.x] += s_sum[threadIdx.x + off];
      s_sq [threadIdx.x] += s_sq [threadIdx.x + off];
    }
    __syncthreads();
  }
  if (threadIdx.x == 0) {
    float mean = s_sum[0] / (float)M;
    float var  = s_sq[0] / (float)M - mean * mean;
    float sc   = g[c] * rsqrtf(var + 1e-5f);
    scale[c] = sc;
    shift[c] = b[c] - mean * sc;
  }
}

// y = x*scale[c] + shift[c] (+res) (relu?) -> f16 and/or f32 outputs
__global__ void bn_apply_k(const float* __restrict__ x, const float* __restrict__ scale,
                           const float* __restrict__ shift, const float* __restrict__ resF,
                           const _Float16* __restrict__ resH, int relu,
                           _Float16* __restrict__ outH, float* __restrict__ outF,
                           size_t total, int C)
{
  size_t idx = (size_t)blockIdx.x * 256 + threadIdx.x;
  if (idx >= total) return;
  int c = (int)(idx % C);
  float v = x[idx] * scale[c] + shift[c];
  if (resF) v += resF[idx];
  if (resH) v += (float)resH[idx];
  if (relu) v = fmaxf(v, 0.f);
  if (outH) outH[idx] = (_Float16)v;
  if (outF) outF[idx] = v;
}

// fused avgpool(L=4) + z@W^T + softmax + shrink + renorm + att@W
__global__ __launch_bounds__(256)
void attention_k(const _Float16* __restrict__ act, const float* __restrict__ W,
                 float* __restrict__ featOut, _Float16* __restrict__ featH)
{
  __shared__ float pooled[DIMZ];
  __shared__ float att[32];
  int n = blockIdx.x;
  for (int c = threadIdx.x; c < DIMZ; c += 256) {
    float s = 0.f;
#pragma unroll
    for (int l = 0; l < 4; ++l) s += (float)act[((size_t)n * 4 + l) * DIMZ + c];
    pooled[c] = 0.25f * s;
  }
  __syncthreads();
  if (threadIdx.x < MEMD) {
    const float* wr = W + threadIdx.x * DIMZ;
    float d = 0.f;
    for (int c = 0; c < DIMZ; ++c) d += pooled[c] * wr[c];
    att[threadIdx.x] = d;
  }
  __syncthreads();
  if (threadIdx.x == 0) {
    float mx = -1e30f;
    for (int j = 0; j < MEMD; ++j) mx = fmaxf(mx, att[j]);
    float s = 0.f;
    for (int j = 0; j < MEMD; ++j) { float e = expf(att[j] - mx); att[j] = e; s += e; }
    float inv = 1.f / s, s2 = 0.f;
    for (int j = 0; j < MEMD; ++j) {
      float a = att[j] * inv;
      a = (a > SHRINK) ? a : 0.f;        // relu(a-s)*a/(|a-s|+eps) == a or 0
      att[j] = a; s2 += a;
    }
    float inv2 = 1.f / fmaxf(s2, 1e-12f);
    for (int j = 0; j < MEMD; ++j) att[j] *= inv2;
  }
  __syncthreads();
  for (int c = threadIdx.x; c < DIMZ; c += 256) {
    float f = 0.f;
    for (int j = 0; j < MEMD; ++j) f += att[j] * W[j * DIMZ + c];
    featOut[(size_t)n * DIMZ + c] = f;
    featH [(size_t)n * DIMZ + c] = (_Float16)f;
  }
}

// per-row: keep 2048 smallest adj entries, normalize by sum|.|, pred = adj@lmY
__global__ __launch_bounds__(256)
void adj_select_k(const float* __restrict__ adj, const float* __restrict__ lmY,
                  float* __restrict__ pred)
{
  __shared__ float vals[N_LM];
  __shared__ unsigned hist[256];
  __shared__ unsigned s_prefix, s_need, s_eq;
  __shared__ float s_sum, s_px, s_py;
  int r = blockIdx.x;
  const float* row = adj + (size_t)r * N_LM;
  for (int j = threadIdx.x; j < N_LM; j += 256) vals[j] = row[j];
  if (threadIdx.x == 0) { s_prefix = 0; s_need = K_ADJ; s_eq = 0;
                          s_sum = 0.f; s_px = 0.f; s_py = 0.f; }
  __syncthreads();
  for (int s = 24; s >= 0; s -= 8) {
    hist[threadIdx.x] = 0;                 // 256 threads == 256 bins
    __syncthreads();
    unsigned pre = s_prefix;
    unsigned hiMask = (s == 24) ? 0u : (0xFFFFFFFFu << (s + 8));
    for (int j = threadIdx.x; j < N_LM; j += 256) {
      unsigned key = ordKey(vals[j]);
      if ((key & hiMask) == pre) atomicAdd(&hist[(key >> s) & 255], 1u);
    }
    __syncthreads();
    if (threadIdx.x == 0) {
      unsigned need = s_need, cum = 0; int bin = 0;
      for (; bin < 256; ++bin) { if (cum + hist[bin] >= need) break; cum += hist[bin]; }
      s_need = need - cum;
      s_prefix = pre | ((unsigned)bin << s);
    }
    __syncthreads();
  }
  unsigned T = s_prefix, need = s_need;
  for (int j = threadIdx.x; j < N_LM; j += 256) {
    float v = vals[j];
    unsigned key = ordKey(v);
    bool keep = (key < T);
    if (!keep && key == T) { unsigned p = atomicAdd(&s_eq, 1u); keep = (p < need); }
    if (keep) {
      atomicAdd(&s_sum, fabsf(v));
      atomicAdd(&s_px, v * lmY[2 * j]);
      atomicAdd(&s_py, v * lmY[2 * j + 1]);
    }
  }
  __syncthreads();
  if (threadIdx.x == 0) {
    float inv = 1.f / fmaxf(s_sum, 1e-12f);
    pred[2 * r]     = s_px * inv;
    pred[2 * r + 1] = s_py * inv;
  }
}

// per-row knn one-hot over 8192 pts (d=2); select 819 smallest d^2 (sqrt monotone)
__global__ __launch_bounds__(256)
void knn_k(const float* __restrict__ lmY, const float* __restrict__ tgY,
           float* __restrict__ out)
{
  __shared__ float d2s[N_TOT];           // 32 KB of the 320 KB/WGP LDS
  __shared__ unsigned hist[256];
  __shared__ unsigned s_prefix, s_need, s_eq;
  int i = blockIdx.x;
  float yi0 = (i < N_LM) ? lmY[2 * i]     : tgY[2 * (i - N_LM)];
  float yi1 = (i < N_LM) ? lmY[2 * i + 1] : tgY[2 * (i - N_LM) + 1];
  for (int j = threadIdx.x; j < N_TOT; j += 256) {
    float yj0 = (j < N_LM) ? lmY[2 * j]     : tgY[2 * (j - N_LM)];
    float yj1 = (j < N_LM) ? lmY[2 * j + 1] : tgY[2 * (j - N_LM) + 1];
    float dx = yi0 - yj0, dy = yi1 - yj1;
    d2s[j] = dx * dx + dy * dy;
  }
  if (threadIdx.x == 0) { s_prefix = 0; s_need = K_KNN; s_eq = 0; }
  __syncthreads();
  for (int s = 24; s >= 0; s -= 8) {
    hist[threadIdx.x] = 0;
    __syncthreads();
    unsigned pre = s_prefix;
    unsigned hiMask = (s == 24) ? 0u : (0xFFFFFFFFu << (s + 8));
    for (int j = threadIdx.x; j < N_TOT; j += 256) {
      unsigned key = ordKey(d2s[j]);
      if ((key & hiMask) == pre) atomicAdd(&hist[(key >> s) & 255], 1u);
    }
    __syncthreads();
    if (threadIdx.x == 0) {
      unsigned need = s_need, cum = 0; int bin = 0;
      for (; bin < 256; ++bin) { if (cum + hist[bin] >= need) break; cum += hist[bin]; }
      s_need = need - cum;
      s_prefix = pre | ((unsigned)bin << s);
    }
    __syncthreads();
  }
  unsigned T = s_prefix, need = s_need;
  float* orow = out + (size_t)i * N_TOT;
  for (int j = threadIdx.x; j < N_TOT; j += 256) {
    unsigned key = ordKey(d2s[j]);
    bool keep = (key < T);
    if (!keep && key == T) { unsigned p = atomicAdd(&s_eq, 1u); keep = (p < need); }
    orow[j] = keep ? 1.0f : 0.0f;
  }
}

static void launch_gemm(const _Float16* A, int lda, const _Float16* Bt, int ldb,
                        float* C, int ldc, int M, int Nc, int Kp, hipStream_t st)
{
  dim3 g((Nc + BN - 1) / BN, M / BM);
  gemm_f16_wmma<<<g, 256, 0, st>>>(A, lda, Bt, ldb, C, ldc, M, Nc, Kp);
}

} // namespace

// ---------------------------------------------------------------------------
extern "C" void kernel_launch(void* const* d_in, const int* in_sizes, int n_in,
                              void* d_out, int out_size, void* d_ws, size_t ws_size,
                              hipStream_t stream)
{
  (void)in_sizes; (void)n_in; (void)out_size; (void)ws_size;

  // inputs, insertion-order flattened
  const float* lm_X = (const float*)d_in[0];
  const float* lm_Y = (const float*)d_in[1];
  const float* tg_X = (const float*)d_in[2];
  const float* tg_Y = (const float*)d_in[3];
  const float* lm_d = (const float*)d_in[4];
  const float* tg_d = (const float*)d_in[5];
  int pi = 6;
  const float* conv1_w = (const float*)d_in[pi++];
  const float* bn1_g   = (const float*)d_in[pi++];
  const float* bn1_b   = (const float*)d_in[pi++];
  const int PLAN_IN [8] = {64, 64, 64, 128, 128, 256, 256, 512};
  const int PLAN_OUT[8] = {64, 64, 128, 128, 256, 256, 512, 512};
  const int STR     [8] = {1, 1, 2, 1, 2, 1, 2, 1};
  struct BP { const float *w1,*g1,*b1,*w2,*g2,*b2,*ws,*gs,*bs; } bp[8];
  for (int k = 0; k < 8; ++k) {
    bp[k].w1 = (const float*)d_in[pi++]; bp[k].g1 = (const float*)d_in[pi++];
    bp[k].b1 = (const float*)d_in[pi++]; bp[k].w2 = (const float*)d_in[pi++];
    bp[k].g2 = (const float*)d_in[pi++]; bp[k].b2 = (const float*)d_in[pi++];
    if (STR[k] != 1 || PLAN_IN[k] != PLAN_OUT[k]) {
      bp[k].ws = (const float*)d_in[pi++]; bp[k].gs = (const float*)d_in[pi++];
      bp[k].bs = (const float*)d_in[pi++];
    } else { bp[k].ws = nullptr; bp[k].gs = nullptr; bp[k].bs = nullptr; }
  }
  const float* memW = (const float*)d_in[pi++];

  // workspace carve
  char* base = (char*)d_ws;
  size_t off = 0;
  auto carve = [&](size_t bytes) -> char* {
    char* p = base + off;
    off = (off + bytes + 255) & ~(size_t)255;
    return p;
  };
  const size_t ACT_ELEMS = (size_t)N_TOT * 32 * 64;        // 16.78M halfs, max act
  const size_t COL_ELEMS = (size_t)N_TOT * 32 * 192;       // 50.3M halfs, max col
  _Float16* actA    = (_Float16*)carve(ACT_ELEMS * 2);
  _Float16* actB    = (_Float16*)carve(ACT_ELEMS * 2);
  _Float16* tmpH    = (_Float16*)carve(ACT_ELEMS * 2);
  _Float16* colBuf  = (_Float16*)carve(COL_ELEMS * 2);
  float*    convOut = (float*)   carve(ACT_ELEMS * 4);     // reused as adj later
  float*    scOut   = (float*)   carve(ACT_ELEMS * 4);
  _Float16* wBuf    = (_Float16*)carve((size_t)1536 * 512 * 2);
  float*    scaleB  = (float*)   carve(512 * 4);
  float*    shiftB  = (float*)   carve(512 * 4);
  float*    adjBuf  = convOut;                             // 4096*4096 f32 fits
  _Float16* featH   = tmpH;                                // 8192*512 halfs fits

  float* outF    = (float*)d_out;
  float* predOut = outF + (size_t)N_TOT * DIMZ;
  float* knnOut  = predOut + (size_t)N_LM * 2;

  auto do_conv = [&](const _Float16* in, int Lin, int Cin, const float* wt,
                     int Cout, int KS, int stride, int pad, float* outBuf) -> int {
    int Lout = (Lin + 2 * pad - KS) / stride + 1;
    int Kp = ((Cin * KS + 31) / 32) * 32;
    int Np = ((Cout + 127) / 128) * 128;
    size_t Mout = (size_t)N_TOT * Lout;
    size_t totc = Mout * Kp;
    im2col_k<<<(unsigned)((totc + 255) / 256), 256, 0, stream>>>(
        in, colBuf, Lin, Cin, Lout, KS, stride, pad, Kp);
    int totw = Np * Kp;
    repack_w_k<<<(totw + 255) / 256, 256, 0, stream>>>(wt, wBuf, Cin, KS, Cout, Kp, Np);
    launch_gemm(colBuf, Kp, wBuf, Kp, outBuf, Cout, (int)Mout, Cout, Kp, stream);
    return Lout;
  };
  auto do_bn = [&](float* x, const float* g, const float* b, size_t M, int C,
                   const float* resF, const _Float16* resH, int relu,
                   _Float16* outH, float* outFb) {
    bn_stats_k<<<C, 256, 0, stream>>>(x, g, b, scaleB, shiftB, (int)M, C);
    size_t tot = M * C;
    bn_apply_k<<<(unsigned)((tot + 255) / 256), 256, 0, stream>>>(
        x, scaleB, shiftB, resF, resH, relu, outH, outFb, tot, C);
  };

  // ---- stem ----
  pack_x_k<<<(N_TOT * 32 + 255) / 256, 256, 0, stream>>>(lm_X, lm_d, tg_X, tg_d, actA);
  _Float16* cur = actA;
  _Float16* nxt = actB;
  do_conv(cur, 32, 1, conv1_w, 64, 3, 1, 1, convOut);
  do_bn(convOut, bn1_g, bn1_b, (size_t)N_TOT * 32, 64, nullptr, nullptr, 1, nxt, nullptr);
  { _Float16* t = cur; cur = nxt; nxt = t; }
  int L = 32, Cin = 64;

  // ---- residual blocks ----
  for (int k = 0; k < 8; ++k) {
    int Cout = PLAN_OUT[k], s = STR[k];
    int Lout = do_conv(cur, L, Cin, bp[k].w1, Cout, 3, s, 1, convOut);
    size_t M = (size_t)N_TOT * Lout;
    do_bn(convOut, bp[k].g1, bp[k].b1, M, Cout, nullptr, nullptr, 1, tmpH, nullptr);
    do_conv(tmpH, Lout, Cout, bp[k].w2, Cout, 3, 1, 1, convOut);
    const float* resF = nullptr; const _Float16* resH = nullptr;
    if (bp[k].ws) {
      do_conv(cur, L, Cin, bp[k].ws, Cout, 1, s, 0, scOut);
      do_bn(scOut, bp[k].gs, bp[k].bs, M, Cout, nullptr, nullptr, 0, nullptr, scOut);
      resF = scOut;
    } else {
      resH = cur;
    }
    do_bn(convOut, bp[k].g2, bp[k].b2, M, Cout, resF, resH, 1, nxt, nullptr);
    { _Float16* t = cur; cur = nxt; nxt = t; }
    Cin = Cout; L = Lout;
  }

  // ---- pool + attention + feature (writes d_out feature + f16 copy) ----
  attention_k<<<N_TOT, 256, 0, stream>>>(cur, memW, outF, featH);

  // ---- adj = f_tg @ f_lm^T (WMMA, Bt = f_lm), then top-k + normalize + pred --
  launch_gemm(featH + (size_t)N_LM * DIMZ, DIMZ, featH, DIMZ,
              adjBuf, N_LM, N_LM, N_LM, DIMZ, stream);
  adj_select_k<<<N_LM, 256, 0, stream>>>(adjBuf, lm_Y, predOut);

  // ---- knn one-hot over pairwise distances ----
  knn_k<<<N_TOT, 256, 0, stream>>>(lm_Y, tg_Y, knnOut);
}
